// SHTCUDA_75660143886827
// MI455X (gfx1250) — compile-verified, matrix operand order checked by reading, not run on hardware
//
#include <hip/hip_runtime.h>
#include <hip/hip_bf16.h>

typedef __attribute__((ext_vector_type(16))) _Float16 v16h;
typedef __attribute__((ext_vector_type(8)))  _Float16 v8h;
typedef __attribute__((ext_vector_type(8)))  float    v8f;

#define NLAT  255
#define NLON  256
#define NPIX  49152
#define LMAX  255
#define MMAX  129
#define BC    256    // BATCH*CHAN
#define KPAD  256    // padded GEMM K (p for pass1, nlat for pass2)
#define C1    258    // 2*MMAX interleaved (fr, fi)
#define C1PAD 288    // padded to 18 tiles of 16 (9 supertiles of 32)
#define LPAD  256

union HV { v16h v; v8h h[2]; };

// Load a 16x32 f16 fragment (A-style: rows x K-contiguous). Works for B too
// when B is stored transposed as [N][K], and for LDS-resident tiles.
// Per ISA 7.12.2: lanes 0-15 hold row=lane, K {0..7,16..23}; lanes 16-31
// hold row=lane-16, K {8..15,24..31}.
static __device__ inline v16h load_frag16(const _Float16* __restrict__ base, int ldk) {
    const int lane = threadIdx.x & 31;
    const int row  = lane & 15;
    const int hi   = (lane >> 4) << 3;       // 0 or 8 halves
    const _Float16* p = base + row * ldk + hi;
    HV u;
    u.h[0] = *reinterpret_cast<const v8h*>(p);       // K 0..7   (or 8..15)
    u.h[1] = *reinterpret_cast<const v8h*>(p + 16);  // K 16..23 (or 24..31)
    return u.v;
}

static __device__ inline v8f wmma(v16h a, v16h b, v8f c) {
    return __builtin_amdgcn_wmma_f32_16x16x32_f16(
               false, a, false, b, (short)0, c, false, false);
}

// ---- Stage 0: gather x by ring index, mask, convert to f16: A1[k][bc][p] ----
__global__ __launch_bounds__(256) void k_gather(
    const float* __restrict__ x, const int* __restrict__ ring_index,
    const float* __restrict__ ring_mask, _Float16* __restrict__ A1)
{
    int idx = blockIdx.x * 256 + threadIdx.x;   // < 255*256*256
    int p   = idx & 255;
    int bc  = (idx >> 8) & 255;
    int k   = idx >> 16;
    int rp  = k * NLON + p;
    float v = x[bc * NPIX + ring_index[rp]] * ring_mask[rp];
    A1[idx] = (_Float16)v;
}

// ---- Stage 0: trig table, transposed for B-fragments: B1t[k][c][p] ----
// c = 2m+n: even -> cos(phase*m), odd -> -sin(phase*m); c>=258 zero pad.
__global__ __launch_bounds__(256) void k_trig(
    const float* __restrict__ phase, _Float16* __restrict__ B1t)
{
    int idx = blockIdx.x * 256 + threadIdx.x;   // < 255*288*256
    int p   = idx & 255;
    int c   = (idx >> 8) % C1PAD;
    int k   = (idx >> 8) / C1PAD;
    float v = 0.0f;
    if (c < C1) {
        float ang = phase[k * NLON + p] * (float)(c >> 1);
        float s, co;
        __sincosf(ang, &s, &co);
        v = (c & 1) ? -s : co;
    }
    B1t[idx] = (_Float16)v;
}

// ---- Stage 0: weights W2[m][l][k] = pct[m,l,k]*w[k], zero-padded k,l=255 ----
__global__ __launch_bounds__(256) void k_wgt(
    const float* __restrict__ pct, const float* __restrict__ w,
    _Float16* __restrict__ W2)
{
    int idx = blockIdx.x * 256 + threadIdx.x;   // < 129*256*256
    int k   = idx & 255;
    int l   = (idx >> 8) & 255;
    int m   = idx >> 16;
    float v = 0.0f;
    if (l < LMAX && k < NLAT)
        v = pct[(m * LMAX + l) * NLAT + k] * w[k];
    W2[idx] = (_Float16)v;
}

// Scatter one 16x16 f32 tile into F[(2m+n)][bc][k] as f16 (K-contiguous for
// pass 2). C-layout per ISA 7.12.2: VGPR r holds (M = r + 8*(lane>=16),
// N = lane&15).
static __device__ inline void store_tile_F(
    const v8f& acc, _Float16* __restrict__ F, int ring, int bc0, int c0)
{
    const int lane = threadIdx.x & 31;
    const int col  = c0 + (lane & 15);          // = 2m+n
    if (col < C1) {
        const int row0 = bc0 + ((lane >> 4) << 3);
        #pragma unroll
        for (int r = 0; r < 8; ++r)
            F[(col * BC + row0 + r) * KPAD + ring] = (_Float16)acc[r];
    }
}

// ---- Pass 1: per-ring DFT. C[bc, c] = A1[ring] x B1t[ring]^T, K=256. ----
// 2x2 register blocking: each wave computes a 32x32 supertile (4 WMMA per
// 2 A + 2 B fragment loads -> 16 FLOP/B from L2). 9 blocks of 8 waves per
// ring cover the 8x9 supertile grid.
__global__ __launch_bounds__(256) void k_dft_wmma(
    const _Float16* __restrict__ A1, const _Float16* __restrict__ B1t,
    _Float16* __restrict__ F)
{
    const int wave = threadIdx.x >> 5;
    const int ring = blockIdx.x / 9;
    const int s    = (blockIdx.x % 9) * 8 + wave;   // 0..71
    const int rs   = s / 9;                         // bc supertile (32 rows)
    const int cs   = s % 9;                         // c  supertile (32 cols)

    const _Float16* aB = A1  + (ring * BC    + rs * 32) * KPAD;
    const _Float16* bB = B1t + (ring * C1PAD + cs * 32) * KPAD;

    v8f acc00 = {}, acc01 = {}, acc10 = {}, acc11 = {};
    #pragma unroll
    for (int kk = 0; kk < 256; kk += 32) {
        v16h a0 = load_frag16(aB + kk,             KPAD);
        v16h a1 = load_frag16(aB + 16 * KPAD + kk, KPAD);
        v16h b0 = load_frag16(bB + kk,             KPAD);
        v16h b1 = load_frag16(bB + 16 * KPAD + kk, KPAD);
        acc00 = wmma(a0, b0, acc00);
        acc01 = wmma(a0, b1, acc01);
        acc10 = wmma(a1, b0, acc10);
        acc11 = wmma(a1, b1, acc11);
    }

    store_tile_F(acc00, F, ring, rs * 32,      cs * 32);
    store_tile_F(acc01, F, ring, rs * 32,      cs * 32 + 16);
    store_tile_F(acc10, F, ring, rs * 32 + 16, cs * 32);
    store_tile_F(acc11, F, ring, rs * 32 + 16, cs * 32 + 16);
}

// ---- Pass 2: Legendre. For each m: C[bc, l] = F[m,n] x W2[m]^T, K=256. ----
// Block = (m, 32-wide l supertile). The 32x256 f16 weight slab (16 KB) is
// staged in LDS once and shared by all 8 waves (ds_load fragments). Each
// wave owns a 32-wide bc supertile and computes 2 bc-tiles x 2 l-tiles for
// both n (8 accumulators, 8 WMMA per 4 A + 2 B fragments). n-pairs stored
// as coalesced float2 matching out's [...,m,n] layout.
__global__ __launch_bounds__(256) void k_leg_wmma(
    const _Float16* __restrict__ F, const _Float16* __restrict__ W2,
    float* __restrict__ out)
{
    __shared__ _Float16 ldsB[32 * KPAD];            // 16 KB

    const int m  = blockIdx.x / 8;
    const int ls = blockIdx.x % 8;                  // l supertile (32 cols)

    // Stage weight slab: 32 rows (l) x 256 (k) f16, coalesced 16B chunks.
    {
        const uint4* src = reinterpret_cast<const uint4*>(
            W2 + (m * LPAD + ls * 32) * KPAD);
        uint4* dst = reinterpret_cast<uint4*>(ldsB);
        #pragma unroll
        for (int i = threadIdx.x; i < (32 * KPAD) / 8; i += 256)
            dst[i] = src[i];
    }
    __syncthreads();

    const int wave = threadIdx.x >> 5;
    const int bc0  = wave * 32;                     // bc supertile per wave

    const _Float16* aR = F + ((m * 2 + 0) * BC + bc0) * KPAD;
    const _Float16* aI = F + ((m * 2 + 1) * BC + bc0) * KPAD;

    v8f acc[2][2][2] = {};                          // [bcT][lT][n]
    #pragma unroll
    for (int kk = 0; kk < 256; kk += 32) {
        v16h b0 = load_frag16(ldsB + kk,             KPAD);
        v16h b1 = load_frag16(ldsB + 16 * KPAD + kk, KPAD);
        v16h a00 = load_frag16(aR + kk,             KPAD);   // bcT=0, n=0
        v16h a01 = load_frag16(aI + kk,             KPAD);   // bcT=0, n=1
        v16h a10 = load_frag16(aR + 16 * KPAD + kk, KPAD);   // bcT=1, n=0
        v16h a11 = load_frag16(aI + 16 * KPAD + kk, KPAD);   // bcT=1, n=1
        acc[0][0][0] = wmma(a00, b0, acc[0][0][0]);
        acc[0][0][1] = wmma(a01, b0, acc[0][0][1]);
        acc[0][1][0] = wmma(a00, b1, acc[0][1][0]);
        acc[0][1][1] = wmma(a01, b1, acc[0][1][1]);
        acc[1][0][0] = wmma(a10, b0, acc[1][0][0]);
        acc[1][0][1] = wmma(a11, b0, acc[1][0][1]);
        acc[1][1][0] = wmma(a10, b1, acc[1][1][0]);
        acc[1][1][1] = wmma(a11, b1, acc[1][1][1]);
    }

    const int lane = threadIdx.x & 31;
    float2* out2 = reinterpret_cast<float2*>(out);
    #pragma unroll
    for (int bcT = 0; bcT < 2; ++bcT) {
        #pragma unroll
        for (int lT = 0; lT < 2; ++lT) {
            const int l = ls * 32 + lT * 16 + (lane & 15);
            if (l < LMAX) {
                const int row0 = bc0 + bcT * 16 + ((lane >> 4) << 3);
                #pragma unroll
                for (int r = 0; r < 8; ++r) {
                    int o = ((row0 + r) * LMAX + l) * MMAX + m;
                    out2[o] = make_float2(acc[bcT][lT][0][r], acc[bcT][lT][1][r]);
                }
            }
        }
    }
}

extern "C" void kernel_launch(void* const* d_in, const int* in_sizes, int n_in,
                              void* d_out, int out_size, void* d_ws, size_t ws_size,
                              hipStream_t stream) {
    const float* x          = (const float*)d_in[0];
    const float* pct        = (const float*)d_in[1];
    const float* w          = (const float*)d_in[2];
    const int*   ring_index = (const int*)  d_in[3];
    const float* ring_mask  = (const float*)d_in[4];
    const float* phase      = (const float*)d_in[5];
    float* out = (float*)d_out;

    // Workspace carving (f16 halves), all L2-resident (~122 MB < 192 MB L2).
    const size_t nA1  = (size_t)NLAT * BC * KPAD;      // 16.7 M halves
    const size_t nB1t = (size_t)NLAT * C1PAD * KPAD;   // 18.8 M halves
    const size_t nW2  = (size_t)MMAX * LPAD * KPAD;    //  8.5 M halves
    char* ws = (char*)d_ws;
    _Float16* A1  = (_Float16*)ws;
    _Float16* B1t = A1  + nA1;
    _Float16* W2  = B1t + nB1t;
    _Float16* F   = W2  + nW2;                         // 16.9 M halves

    // Stage 0: operand preparation (same stream serializes dependencies).
    k_gather<<<(NLAT * BC * KPAD) / 256, 256, 0, stream>>>(x, ring_index, ring_mask, A1);
    k_trig  <<<(NLAT * C1PAD * KPAD) / 256, 256, 0, stream>>>(phase, B1t);
    k_wgt   <<<(MMAX * LPAD * KPAD) / 256, 256, 0, stream>>>(pct, w, W2);

    // Pass 1: per-ring DFT (255 rings x 9 blocks of 8 waves, 32x32/wave).
    k_dft_wmma<<<NLAT * 9, 256, 0, stream>>>(A1, B1t, F);

    // Pass 2: Legendre contraction (129 m x 8 l-supertiles).
    k_leg_wmma<<<MMAX * 8, 256, 0, stream>>>(F, W2, out);
}
